// GRANGER_54099408060420
// MI455X (gfx1250) — compile-verified, matrix operand order checked by reading, not run on hardware
//
#include <hip/hip_runtime.h>
#include <hip/hip_bf16.h>

// ---------------------------------------------------------------------------
// GRANGER-style model for MI455X (gfx1250, wave32, WMMA).
// Sizes: B=64, T=110, P=64, H=256, K=16, L=T-10=100, 3H=768.
// Outputs (f32, concatenated): pred[P,B,L,1] | log_var[B,H] | mu[B,H] |
//                              mean_[B,P] | disp_[B,P]
// ---------------------------------------------------------------------------

typedef __attribute__((ext_vector_type(16))) _Float16 v16h;
typedef __attribute__((ext_vector_type(8)))  _Float16 v8h;
typedef __attribute__((ext_vector_type(8)))  float    v8f;

#define NB   64
#define NT   110
#define NP   64
#define NH   256
#define NK   16
#define NL   100
#define G3H  768

// out offsets (elements)
#define OFF_PRED 0
#define OFF_LV   409600
#define OFF_MU   425984
#define OFF_MEAN 442368
#define OFF_DISP 446464

// ws layout (bytes)
#define WS_Z_OFF    0            // 64*256 f32 = 65536 B
#define WS_WIHF_OFF 65536        // 64*48*512 halves = 3145728 B
#define WS_WHHF_OFF 3211264      // 64*48*8*512 halves = 25165824 B

#define NIH_HALVES (64 * 48 * 512)        // 1,572,864
#define NHH_HALVES (64 * 48 * 8 * 512)    // 12,582,912
#define PREP_TOTAL (NIH_HALVES + NHH_HALVES)

__device__ __forceinline__ float sigm(float x) { return 1.0f / (1.0f + __expf(-x)); }

// ---------------------------------------------------------------------------
// Prep: repack W_ih_nets / W_hh_nets (f32) into f16 WMMA B-fragment layout.
// Fragment = [lane(32)][16 halves] (32 B contiguous per lane).
// A/B half mapping: halves 0..7 -> K = ks*32 + hi*8 + i
//                   halves 8..15 -> K = ks*32 + 16 + hi*8 + (i-8)
// where hi = lane>>4, N = nt*16 + (lane&15).
// W_ih (K=16) is zero-padded to K=32 (single ks).
// ---------------------------------------------------------------------------
__global__ void granger_prep_kernel(const float* __restrict__ Wih,
                                    const float* __restrict__ Whh,
                                    _Float16* __restrict__ wihF,
                                    _Float16* __restrict__ whhF) {
    int idx = blockIdx.x * 256 + threadIdx.x;
    if (idx < NIH_HALVES) {
        int within = idx & 511;
        int frag   = idx >> 9;
        int nt = frag % 48;
        int p  = frag / 48;
        int i    = within & 15;
        int lane = within >> 4;
        int n  = nt * 16 + (lane & 15);
        int hi = lane >> 4;
        int k  = hi * 8 + ((i < 8) ? i : 16 + (i - 8));
        float v = (k < NK) ? Wih[(p * G3H + n) * NK + k] : 0.0f;
        wihF[idx] = (_Float16)v;
    } else if (idx < PREP_TOTAL) {
        int g = idx - NIH_HALVES;
        int within = g & 511;
        int frag   = g >> 9;
        int ks = frag & 7;
        int nt = (frag >> 3) % 48;
        int p  = frag / (8 * 48);
        int i    = within & 15;
        int lane = within >> 4;
        int n  = nt * 16 + (lane & 15);
        int hi = lane >> 4;
        int k  = ks * 32 + hi * 8 + ((i < 8) ? i : 16 + (i - 8));
        whhF[g] = (_Float16)Whh[(p * G3H + n) * NH + k];
    }
}

// ---------------------------------------------------------------------------
// Encoder: gru_left (10 steps) -> gru_1 (1 step, h0=0) -> heads.
// Single 1024-thread workgroup, fp32 VALU (tiny: ~0.3 GFLOP).
// Dynamic LDS: hA[16384] | hB[16384] | xs[4096] floats = 147456 B.
// ---------------------------------------------------------------------------
__launch_bounds__(1024)
__global__ void granger_encoder_kernel(
    const float* __restrict__ X,
    const float* __restrict__ WihL, const float* __restrict__ WhhL,
    const float* __restrict__ bihL, const float* __restrict__ bhhL,
    const float* __restrict__ Wih1, const float* __restrict__ Whh1,
    const float* __restrict__ bih1, const float* __restrict__ bhh1,
    const float* __restrict__ Wmu,  const float* __restrict__ bmu,
    const float* __restrict__ Wstd, const float* __restrict__ bstd,
    const float* __restrict__ Wmean,const float* __restrict__ bmean,
    const float* __restrict__ Wdisp,const float* __restrict__ bdisp,
    const float* __restrict__ znoise,
    float* __restrict__ z_ws, float* __restrict__ out) {
    extern __shared__ float enc_smem[];
    float* hA = enc_smem;            // 64 x 256
    float* hB = enc_smem + 16384;    // 64 x 256
    float* xs = enc_smem + 32768;    // 64 x 64
    const int tid = threadIdx.x;

    for (int e = tid; e < 16384; e += 1024) hA[e] = 0.0f;
    __syncthreads();

    float* cur = hA;
    float* nxt = hB;
    // gru_left over X[:, 0:10, :]  (== X_pad[:, 1:11, :])
    for (int t = 0; t < 10; ++t) {
        for (int e = tid; e < 4096; e += 1024)
            xs[e] = X[((e >> 6) * NT + t) * NP + (e & 63)];
        __syncthreads();
        for (int o = tid; o < 16384; o += 1024) {
            int b = o >> 8, k = o & 255;
            const float* xb = xs + b * 64;
            const float* hb = cur + b * 256;
            float xr = 0.f, xz = 0.f, xn = 0.f;
            for (int j = 0; j < 64; ++j) {
                float xv = xb[j];
                xr += xv * WihL[k * 64 + j];
                xz += xv * WihL[(256 + k) * 64 + j];
                xn += xv * WihL[(512 + k) * 64 + j];
            }
            float hr = 0.f, hz = 0.f, hn = 0.f;
            for (int j = 0; j < 256; ++j) {
                float hv = hb[j];
                hr += hv * WhhL[k * 256 + j];
                hz += hv * WhhL[(256 + k) * 256 + j];
                hn += hv * WhhL[(512 + k) * 256 + j];
            }
            float r  = sigm(xr + bihL[k]       + hr + bhhL[k]);
            float zg = sigm(xz + bihL[256 + k] + hz + bhhL[256 + k]);
            float n  = tanhf(xn + bihL[512 + k] + r * (hn + bhhL[512 + k]));
            nxt[o] = (1.0f - zg) * n + zg * hb[k];
        }
        __syncthreads();
        float* tmp = cur; cur = nxt; nxt = tmp;
    }
    // gru_1: x = h_left (cur), h = 0 -> h1 in nxt
    for (int o = tid; o < 16384; o += 1024) {
        int b = o >> 8, k = o & 255;
        const float* hl = cur + b * 256;
        float gr = 0.f, gz = 0.f, gn = 0.f;
        for (int j = 0; j < 256; ++j) {
            float v = hl[j];
            gr += v * Wih1[k * 256 + j];
            gz += v * Wih1[(256 + k) * 256 + j];
            gn += v * Wih1[(512 + k) * 256 + j];
        }
        float r  = sigm(gr + bih1[k]       + bhh1[k]);
        float zg = sigm(gz + bih1[256 + k] + bhh1[256 + k]);
        float n  = tanhf(gn + bih1[512 + k] + r * bhh1[512 + k]);
        nxt[o] = (1.0f - zg) * n;
    }
    __syncthreads();
    const float* h1 = nxt;
    // heads: mu, log_var, z; mean_, disp_
    for (int o = tid; o < 16384; o += 1024) {
        int b = o >> 8, k = o & 255;
        const float* hb = h1 + b * 256;
        float mu = 0.f, lv = 0.f;
        for (int j = 0; j < 256; ++j) {
            float v = hb[j];
            mu += v * Wmu[k * 256 + j];
            lv += v * Wstd[k * 256 + j];
        }
        mu += bmu[k];
        lv += bstd[k];
        out[OFF_MU + o] = mu;
        out[OFF_LV + o] = lv;
        z_ws[o] = mu + __expf(0.5f * lv) * znoise[o];
    }
    for (int o = tid; o < 4096; o += 1024) {
        int b = o >> 6, j = o & 63;
        const float* hb = h1 + b * 256;
        float m = 0.f, d = 0.f;
        for (int i = 0; i < 256; ++i) {
            float v = hb[i];
            m += v * Wmean[j * 256 + i];
            d += v * Wdisp[j * 256 + i];
        }
        m += bmean[j];
        d += bdisp[j];
        out[OFF_MEAN + o] = fminf(fmaxf(__expf(m), 1e-5f), 1e6f);
        out[OFF_DISP + o] = fminf(fmaxf(log1pf(__expf(d)), 1e-4f), 1e4f);
    }
}

// ---------------------------------------------------------------------------
// Decoder helpers: WMMA fragment loads.
// ---------------------------------------------------------------------------
__device__ __forceinline__ v16h combine16(v8h lo, v8h hi) {
    union { v16h v; v8h h[2]; } u;
    u.h[0] = lo; u.h[1] = hi;
    return u.v;
}

// A-fragment (16x32 f16) from row-major LDS matrix (rows=M, stride in halves).
__device__ __forceinline__ v16h load_afrag(const _Float16* base, int stride,
                                           int mt, int ks, int lane) {
    int lo = lane & 15;
    int hi = (lane >> 4) & 1;
    const _Float16* rp = base + (mt * 16 + lo) * stride + ks * 32 + hi * 8;
    v8h a0 = *(const v8h*)(rp);
    v8h a1 = *(const v8h*)(rp + 16);
    return combine16(a0, a1);
}

__device__ __forceinline__ v8f wmma_f16(v16h a, v16h b, v8f c) {
    return __builtin_amdgcn_wmma_f32_16x16x32_f16(false, a, false, b,
                                                  (short)0, c, false, false);
}

// ---------------------------------------------------------------------------
// Decoder: per-series GRU (input K=16, hidden 256, 100 steps) + ReLU + linear.
// One workgroup per p (64 WGs x 256 threads = 8 waves).
// Per step each wave owns gate-triplet groups (mt, kt): tiles r(kt), z(16+kt),
// n(32+kt) + separate i_n, so gate math happens entirely in registers on the
// C-fragment layout; h is double-buffered f16 in LDS (stride 264 for banks).
// Weights stream from L2 in pre-swizzled fragments (32 B/lane).
// Dynamic LDS halves: hA[0..16896) hB[16896..33792) x[33792..36352)
// then floats: bih[768] bhh[768] wlin[256] predb[64]  -> 80128 B total.
// ---------------------------------------------------------------------------
__launch_bounds__(256)
__global__ void granger_decoder_kernel(
    const float* __restrict__ X, const int* __restrict__ conn,
    const _Float16* __restrict__ wihF, const _Float16* __restrict__ whhF,
    const float* __restrict__ bih, const float* __restrict__ bhh,
    const float* __restrict__ wlin, const float* __restrict__ blin,
    const float* __restrict__ z_ws, float* __restrict__ out) {
    extern __shared__ _Float16 dec_smem[];
    _Float16* hbuf0 = dec_smem;            // 64 x 264
    _Float16* hbuf1 = dec_smem + 16896;    // 64 x 264
    _Float16* x_lds = dec_smem + 33792;    // 64 x 40 (cols 16..31 zero pad)
    float* fbase  = (float*)(dec_smem + 36352);
    float* bih_l  = fbase;                 // 768
    float* bhh_l  = fbase + 768;           // 768
    float* wl_l   = fbase + 1536;          // 256
    float* predb  = fbase + 1792;          // 64
    __shared__ int conn_l[NK];

    const int p    = blockIdx.x;
    const int tid  = threadIdx.x;
    const int lane = tid & 31;
    const int wave = tid >> 5;
    const int lo   = lane & 15;
    const int rsel = (lane >= 16) ? 8 : 0;

    // preload biases / linear head / connectivity
    for (int i = tid; i < G3H; i += 256) {
        bih_l[i] = bih[p * G3H + i];
        bhh_l[i] = bhh[p * G3H + i];
    }
    if (tid < NH) wl_l[tid] = wlin[p * NH + tid];
    if (tid < NK) conn_l[tid] = conn[p * NK + tid];
    // h0 = z (broadcast across p), zero x pad columns
    for (int e = tid; e < NB * NH; e += 256) {
        int b = e >> 8, k = e & 255;
        hbuf0[b * 264 + k] = (_Float16)z_ws[e];
    }
    for (int e = tid; e < NB * 24; e += 256) {
        int b = e / 24, c = 16 + (e % 24);
        x_lds[b * 40 + c] = (_Float16)0.0f;
    }
    __syncthreads();

    const float blin_p = blin[p];
    _Float16* hcur = hbuf0;
    _Float16* hnxt = hbuf1;

    for (int s = 0; s < NL; ++s) {
        // ---- phase 1: stage x_t (gathered), zero pred accumulator ----
        if (s == 0) {
            for (int e = tid; e < NB * NK; e += 256) {
                int b = e >> 4, j = e & 15;
                x_lds[b * 40 + j] = (_Float16)0.0f;
            }
        } else {
            int t = 9 + s;  // X rows 10..108
            for (int e = tid; e < NB * NK; e += 256) {
                int b = e >> 4, j = e & 15;
                x_lds[b * 40 + j] = (_Float16)X[(b * NT + t) * NP + conn_l[j]];
            }
        }
        if (tid < NB) predb[tid] = 0.0f;
        __syncthreads();

        // ---- phase 2: WMMA gate triplets + in-register GRU cell ----
        for (int g = wave; g < 64; g += 8) {
            int mt = g & 3;
            int kt = g >> 2;
            // x contributions (K=16 padded to 32, single k-step)
            v16h ax  = load_afrag(x_lds, 40, mt, 0, lane);
            v16h bxr = *(const v16h*)(wihF + (((p * 48) + kt)      << 9) + (lane << 4));
            v16h bxz = *(const v16h*)(wihF + (((p * 48) + 16 + kt) << 9) + (lane << 4));
            v16h bxn = *(const v16h*)(wihF + (((p * 48) + 32 + kt) << 9) + (lane << 4));
            v8f zero = {};
            v8f ar  = wmma_f16(ax, bxr, zero);   // i_r (pre-bias)
            v8f az  = wmma_f16(ax, bxz, zero);   // i_z
            v8f ain = wmma_f16(ax, bxn, zero);   // i_n (kept separate!)
            v8f ahn = zero;                      // h_n
#pragma unroll
            for (int ks = 0; ks < 8; ++ks) {
                v16h a  = load_afrag(hcur, 264, mt, ks, lane);
                const _Float16* wb = whhF + (lane << 4);
                v16h br = *(const v16h*)(wb + (((((p * 48) + kt)      << 3) + ks) << 9));
                v16h bz = *(const v16h*)(wb + (((((p * 48) + 16 + kt) << 3) + ks) << 9));
                v16h bn = *(const v16h*)(wb + (((((p * 48) + 32 + kt) << 3) + ks) << 9));
                ar  = wmma_f16(a, br, ar);   // fuse h_r into i_r
                az  = wmma_f16(a, bz, az);   // fuse h_z into i_z
                ahn = wmma_f16(a, bn, ahn);  // h_n stays separate for r*h_n
            }
            // gate math directly on C layout: row = mt*16 + rsel + i, col = k
            int k = kt * 16 + lo;
            float bihr = bih_l[k],       bhhr = bhh_l[k];
            float bihz = bih_l[256 + k], bhhz = bhh_l[256 + k];
            float bihn = bih_l[512 + k], bhhn = bhh_l[512 + k];
            float wl = wl_l[k];
#pragma unroll
            for (int i = 0; i < 8; ++i) {
                int row = mt * 16 + rsel + i;
                float r    = sigm(ar[i] + bihr + bhhr);
                float zg   = sigm(az[i] + bihz + bhhz);
                float n    = tanhf(ain[i] + bihn + r * (ahn[i] + bhhn));
                float hold = (float)hcur[row * 264 + k];
                float hnew = (1.0f - zg) * n + zg * hold;
                hnxt[row * 264 + k] = (_Float16)hnew;
                // pred contribution: relu(h)*W_lin, reduce over 16 cols/half-wave
                float c = fmaxf(hnew, 0.0f) * wl;
                c += __shfl_xor(c, 1);
                c += __shfl_xor(c, 2);
                c += __shfl_xor(c, 4);
                c += __shfl_xor(c, 8);
                if (lo == 0) atomicAdd(&predb[row], c);
            }
        }
        __syncthreads();

        // ---- phase 3: emit pred ----
        if (tid < NB)
            out[OFF_PRED + (p * NB + tid) * NL + s] = predb[tid] + blin_p;
        __syncthreads();

        _Float16* tmp = hcur; hcur = hnxt; hnxt = tmp;
    }
}

// ---------------------------------------------------------------------------
extern "C" void kernel_launch(void* const* d_in, const int* in_sizes, int n_in,
                              void* d_out, int out_size, void* d_ws, size_t ws_size,
                              hipStream_t stream) {
    (void)in_sizes; (void)n_in; (void)out_size; (void)ws_size;
    const float* X        = (const float*)d_in[0];
    const int*   conn     = (const int*)d_in[1];
    const float* WihL     = (const float*)d_in[2];
    const float* WhhL     = (const float*)d_in[3];
    const float* bihL     = (const float*)d_in[4];
    const float* bhhL     = (const float*)d_in[5];
    const float* Wih1     = (const float*)d_in[6];
    const float* Whh1     = (const float*)d_in[7];
    const float* bih1     = (const float*)d_in[8];
    const float* bhh1     = (const float*)d_in[9];
    const float* Wmu      = (const float*)d_in[10];
    const float* bmu      = (const float*)d_in[11];
    const float* Wstd     = (const float*)d_in[12];
    const float* bstd     = (const float*)d_in[13];
    const float* Wmean    = (const float*)d_in[14];
    const float* bmean    = (const float*)d_in[15];
    const float* Wdisp    = (const float*)d_in[16];
    const float* bdisp    = (const float*)d_in[17];
    const float* Wih_nets = (const float*)d_in[18];
    const float* Whh_nets = (const float*)d_in[19];
    const float* bih_nets = (const float*)d_in[20];
    const float* bhh_nets = (const float*)d_in[21];
    const float* W_lin    = (const float*)d_in[22];
    const float* b_lin    = (const float*)d_in[23];
    const float* znoise   = (const float*)d_in[24];

    float* out = (float*)d_out;
    float*     z_ws = (float*)d_ws;
    _Float16*  wihF = (_Float16*)((char*)d_ws + WS_WIHF_OFF);
    _Float16*  whhF = (_Float16*)((char*)d_ws + WS_WHHF_OFF);

    // 1) repack per-series GRU weights into WMMA fragment layout (f16)
    granger_prep_kernel<<<dim3(PREP_TOTAL / 256), dim3(256), 0, stream>>>(
        Wih_nets, Whh_nets, wihF, whhF);

    // 2) encoder -> z (ws), mu/log_var/mean_/disp_ (out)
    granger_encoder_kernel<<<dim3(1), dim3(1024), 147456, stream>>>(
        X, WihL, WhhL, bihL, bhhL, Wih1, Whh1, bih1, bhh1,
        Wmu, bmu, Wstd, bstd, Wmean, bmean, Wdisp, bdisp, znoise, z_ws, out);

    // 3) decoder: 64 series-parallel GRUs, WMMA recurrence -> pred (out)
    granger_decoder_kernel<<<dim3(NP), dim3(256), 80128, stream>>>(
        X, conn, wihF, whhF, bih_nets, bhh_nets, W_lin, b_lin, z_ws, out);
}
